// Attention3D_13864154432270
// MI455X (gfx1250) — compile-verified, matrix operand order checked by reading, not run on hardware
//
#include <hip/hip_runtime.h>
#include <hip/hip_bf16.h>
#include <math.h>

typedef _Float16 h16v __attribute__((ext_vector_type(16)));
typedef _Float16 h8v  __attribute__((ext_vector_type(8)));
typedef float    f8v  __attribute__((ext_vector_type(8)));
typedef unsigned int u4v __attribute__((ext_vector_type(4)));
typedef int      i4v __attribute__((ext_vector_type(4)));
typedef int      i8v __attribute__((ext_vector_type(8)));

#define DIMC 768
#define NHEAD 12
#define HD 64
#define BATCH 2
#define DDIM 8
#define HDIM 16
#define WDIM 16
#define NTOK 2048          // DDIM*HDIM*WDIM
#define BH 24              // BATCH*NHEAD
#define ROWS 4096          // BATCH*NTOK
#define SCALE 0.125f       // 64^-0.5

#if defined(__has_builtin)
#  if __has_builtin(__builtin_amdgcn_tensor_load_to_lds)
#    define HAVE_TDM 1
#  else
#    define HAVE_TDM 0
#  endif
#  if __has_builtin(__builtin_amdgcn_s_wait_tensorcnt)
#    define WAIT_TENSOR(n) __builtin_amdgcn_s_wait_tensorcnt(n)
#  else
#    define WAIT_TENSOR(n) asm volatile("s_wait_tensorcnt " #n ::: "memory")
#  endif
#else
#  define HAVE_TDM 0
#  define WAIT_TENSOR(n) asm volatile("s_wait_tensorcnt " #n ::: "memory")
#endif

// ---------------------------------------------------------------- helpers

__device__ __forceinline__ f8v wmma32(h16v a, h16v b, f8v c) {
  // D = A(16x32 f16) * B(32x16 f16) + C(16x16 f32)
  return __builtin_amdgcn_wmma_f32_16x16x32_f16(false, a, false, b,
                                                (short)0, c, false, false);
}

// WMMA A/B fragment from a row-major f16 matrix (global or LDS).
// Element j <-> K = (j%8) + 8*half + 16*(j/8)  => two contiguous 16B loads.
__device__ __forceinline__ h16v frag_load(const _Float16* rowbase, int kc, int half) {
  h8v a = *(const h8v*)(rowbase + kc + 8 * half);
  h8v b = *(const h8v*)(rowbase + kc + 16 + 8 * half);
  return __builtin_shufflevector(a, b, 0,1,2,3,4,5,6,7,8,9,10,11,12,13,14,15);
}

#if HAVE_TDM
// Flat LDS address -> LDS byte offset (aperture keeps offset in addr[31:0]).
__device__ __forceinline__ unsigned lds_off(const void* p) {
  return (unsigned)(unsigned long long)(uintptr_t)p;
}

// Issue a 2-D TDM load: tile (tile_d0 x tile_d1) of 2-byte elements from a
// row-major tensor (row length tensor_d0, row stride stride0) into LDS.
// D# packing per CDNA5 ISA 8.3/8.4: group0 {count=1, lds_addr, global_addr,
// type=2}; group1 {data_size=1(2B), tensor dims, tile dims, dim0 stride}.
__device__ __forceinline__ void tdm_load_2d(unsigned lds_byte_off, const void* gptr,
                                            unsigned tensor_d0, unsigned tensor_d1,
                                            unsigned tile_d0, unsigned tile_d1,
                                            unsigned stride0) {
  const unsigned long long ga = (unsigned long long)(uintptr_t)gptr;
  u4v g0;
  g0[0] = 1u;                                            // count=1 (valid user D#)
  g0[1] = lds_byte_off;                                  // lds_addr
  g0[2] = (unsigned)(ga & 0xffffffffu);                  // global_addr[31:0]
  g0[3] = (unsigned)((ga >> 32) & 0x01ffffffu) | 0x80000000u;  // addr[56:32] | type=2
  i8v g1;
  g1[0] = (int)(1u << 16);                               // data_size=1 => 2 bytes
  g1[1] = (int)((tensor_d0 & 0xffffu) << 16);            // tensor_dim0[15:0] @ bits 63:48
  g1[2] = (int)((tensor_d0 >> 16) | ((tensor_d1 & 0xffffu) << 16)); // dim0 hi | dim1 lo
  g1[3] = (int)((tensor_d1 >> 16) | (tile_d0 << 16));    // dim1 hi | tile_dim0
  g1[4] = (int)tile_d1;                                  // tile_dim1 (tile_dim2=0)
  g1[5] = (int)stride0;                                  // tensor_dim0_stride[31:0]
  g1[6] = 0;                                             // stride0 hi | stride1 lo
  g1[7] = 0;
  const i4v z4 = {0, 0, 0, 0};
#if __clang_major__ >= 23
  const i8v z8 = {0, 0, 0, 0, 0, 0, 0, 0};
  __builtin_amdgcn_tensor_load_to_lds(g0, g1, z4, z4, z8, 0);
#else
  __builtin_amdgcn_tensor_load_to_lds(g0, g1, z4, z4, 0);
#endif
}
#endif  // HAVE_TDM

// ---------------------------------------------------------------- prep

__global__ void k_cvt(const float* __restrict__ src, _Float16* __restrict__ dst, int n) {
  int i = blockIdx.x * blockDim.x + threadIdx.x;
  if (i < n) dst[i] = (_Float16)src[i];
}

// wT[n][k] = (f16) w[k][n]   (w is [K][Ncol] row-major)
__global__ void k_transpose(const float* __restrict__ w, _Float16* __restrict__ wt,
                            int K, int Ncol) {
  long i = (long)blockIdx.x * blockDim.x + threadIdx.x;
  if (i >= (long)K * Ncol) return;
  int n = (int)(i % Ncol), k = (int)(i / Ncol);
  wt[(long)n * K + k] = (_Float16)w[(long)k * Ncol + n];
}

// ---------------------------------------------------------------- qkv GEMM
// C(4096 x 2304) = A_f16 @ BT_f16^T + bias; one wave per 32x64 tile
// (2 A-frags x 4 B-frags -> 8 WMMA per 12 loads). Epilogue scatters q/k/vT.

__global__ void k_gemm_qkv(const _Float16* __restrict__ A, const _Float16* __restrict__ BT,
                           const float* __restrict__ bias,
                           _Float16* __restrict__ qw, _Float16* __restrict__ kw,
                           _Float16* __restrict__ vT) {
  const int lane = threadIdx.x & 31, half = lane >> 4, lo = lane & 15;
  const int wv = threadIdx.x >> 5;
  const int tile = blockIdx.x * 8 + wv;
  const int TN = 2304 / 64;                 // 36 column strips
  const int tm = tile / TN, tn = tile % TN;

  const _Float16* arow0 = A + (long)(tm * 32 + lo) * DIMC;
  const _Float16* arow1 = A + (long)(tm * 32 + 16 + lo) * DIMC;
  const _Float16* brow[4];
  #pragma unroll
  for (int j = 0; j < 4; j++) brow[j] = BT + (long)(tn * 64 + j * 16 + lo) * DIMC;

  f8v acc[2][4] = {};
  #pragma unroll 2
  for (int kc = 0; kc < DIMC; kc += 32) {
    const h16v a0 = frag_load(arow0, kc, half);
    const h16v a1 = frag_load(arow1, kc, half);
    #pragma unroll
    for (int j = 0; j < 4; j++) {
      const h16v bf = frag_load(brow[j], kc, half);
      acc[0][j] = wmma32(a0, bf, acc[0][j]);
      acc[1][j] = wmma32(a1, bf, acc[1][j]);
    }
  }

  const int cb = tn * 64;                   // 64-aligned: one (part, head) strip
  const int part = cb / DIMC;               // 0:q 1:k 2:v
  const int hh = (cb % DIMC) / HD;
  #pragma unroll
  for (int j = 0; j < 4; j++) {
    const int d = j * 16 + lo;
    const float bs = bias[cb + j * 16 + lo];
    #pragma unroll
    for (int i = 0; i < 2; i++) {
      #pragma unroll
      for (int r = 0; r < 8; r++) {
        const int row = tm * 32 + i * 16 + r + 8 * half;
        const int b = row / NTOK, n = row % NTOK;
        const int bh = b * NHEAD + hh;
        const _Float16 hv = (_Float16)(acc[i][j][r] + bs);
        if (part == 0)      qw[((long)bh * NTOK + n) * HD + d] = hv;
        else if (part == 1) kw[((long)bh * NTOK + n) * HD + d] = hv;
        else                vT[((long)bh * HD + d) * NTOK + n] = hv;
      }
    }
  }
}

// ---------------------------------------------------------------- rel-pos dots
// rel_d[bh][n][kd] = q[bh][n][:] . rel_pos_d[qd-kd+D-1][:]   (and h, w)

__global__ void k_rel(const _Float16* __restrict__ q,
                      const float* __restrict__ rpd, const float* __restrict__ rph,
                      const float* __restrict__ rpw,
                      float* __restrict__ rd, float* __restrict__ rh,
                      float* __restrict__ rw) {
  long t = (long)blockIdx.x * blockDim.x + threadIdx.x;
  if (t >= (long)BH * NTOK * 40) return;
  const int j = (int)(t % 40);
  const long bn = t / 40;
  const int n = (int)(bn % NTOK);
  const int qd = n >> 8, qh = (n >> 4) & 15, qw_ = n & 15;
  const _Float16* qv = q + bn * HD;
  const float* tab;
  if (j < 8)       tab = rpd + (long)(qd  - j        + DDIM - 1) * HD;
  else if (j < 24) tab = rph + (long)(qh  - (j - 8)  + HDIM - 1) * HD;
  else             tab = rpw + (long)(qw_ - (j - 24) + WDIM - 1) * HD;
  float s = 0.f;
  #pragma unroll 8
  for (int i = 0; i < HD; i++) s += (float)qv[i] * tab[i];
  if (j < 8)       rd[bn * 8  + j]        = s;
  else if (j < 24) rh[bn * 16 + (j - 8)]  = s;
  else             rw[bn * 16 + (j - 24)] = s;
}

// ---------------------------------------------------------------- flash attention
// grid (BH, NTOK/128); block 256 = 8 waves; each wave owns 16 query rows.
// K/V tiles (4KB each) are staged into LDS once per block via the Tensor Data
// Mover (double-buffered, TENSORcnt-tracked); all 8 waves consume from LDS.

__global__ void k_attn(const _Float16* __restrict__ q, const _Float16* __restrict__ kk,
                       const _Float16* __restrict__ vT,
                       const float* __restrict__ rd, const float* __restrict__ rh,
                       const float* __restrict__ rw,
                       _Float16* __restrict__ aout) {
  __shared__ __align__(16) _Float16 ktile[2][32][HD];   // 32 keys x 64 hd
  __shared__ __align__(16) _Float16 vtile[2][HD][32];   // 64 hd rows x 32 keys
  __shared__ __align__(16) _Float16 pbuf[8][16][32];    // per-wave P relayout
  const int lane = threadIdx.x & 31, half = lane >> 4, lo = lane & 15;
  const int wv = threadIdx.x >> 5;
  const int bh = blockIdx.x;
  const int qbase = blockIdx.y * 128 + wv * 16;
  const int b = bh / NHEAD, hh = bh % NHEAD;

  auto stage = [&](int kt, int buf) {
    const int k0 = kt * 32;
#if HAVE_TDM
    if (wv == 0) {
      // K: 32 rows x 64 cols, contiguous rows of 64 f16
      tdm_load_2d(lds_off(&ktile[buf][0][0]),
                  kk + ((long)bh * NTOK + k0) * HD, HD, NTOK, HD, 32, HD);
      // V^T: 64 rows x 32 cols, row stride NTOK f16
      tdm_load_2d(lds_off(&vtile[buf][0][0]),
                  vT + (long)bh * HD * NTOK + k0, NTOK, HD, 32, HD, NTOK);
    }
#else
    const int t = threadIdx.x;                          // 256 threads x 16B each
    *((h8v*)&ktile[buf][0][0] + t) =
        *((const h8v*)(kk + ((long)bh * NTOK + k0) * HD) + t);
    *((h8v*)&vtile[buf][0][0] + t) =
        *(const h8v*)(vT + ((long)bh * HD + (t >> 2)) * NTOK + k0 + (t & 3) * 8);
#endif
  };

  // q fragments (row m = qbase+lo, hd chunks 0..31 / 32..63), reused all k-tiles
  const _Float16* qrow = q + ((long)bh * NTOK + qbase + lo) * HD;
  const h16v a0 = frag_load(qrow, 0, half);
  const h16v a1 = frag_load(qrow, 32, half);

  f8v acc0 = {}, acc1 = {}, acc2 = {}, acc3 = {};
  float mi[8], li[8], rwv[8];
  #pragma unroll
  for (int r = 0; r < 8; r++) {
    mi[r] = -1e30f; li[r] = 0.f;
    rwv[r] = rw[((long)bh * NTOK + qbase + r + 8 * half) * 16 + lo];
  }

  stage(0, 0);
  const int NT = NTOK / 32;                             // 64 key tiles
  for (int kt = 0; kt < NT; ++kt) {
    const int buf = kt & 1;
    if (kt + 1 < NT) stage(kt + 1, buf ^ 1);            // async prefetch next tile
#if HAVE_TDM
    if (wv == 0) {
      if (kt + 1 < NT) WAIT_TENSOR(2);                  // current tile done (in-order)
      else             WAIT_TENSOR(0);
    }
#endif
    __syncthreads();

    const int k0 = kt * 32;
    f8v s[2];
    #pragma unroll
    for (int hs = 0; hs < 2; ++hs) {
      const _Float16* krow = &ktile[buf][hs * 16 + lo][0];
      f8v sv = {};
      sv = wmma32(a0, frag_load(krow, 0, half), sv);
      sv = wmma32(a1, frag_load(krow, 32, half), sv);
      const int kb = k0 + hs * 16;
      const int kd = kb >> 8, khh = (kb >> 4) & 15;     // uniform over 16-key tile
      #pragma unroll
      for (int r = 0; r < 8; r++) {
        const long qi = (long)bh * NTOK + qbase + r + 8 * half;
        sv[r] = sv[r] * SCALE + rd[qi * 8 + kd] + rh[qi * 16 + khh] + rwv[r];
      }
      s[hs] = sv;
    }
    // online softmax over these 32 keys (row reductions across 16-lane groups)
    #pragma unroll
    for (int r = 0; r < 8; r++) {
      float mx = fmaxf(s[0][r], s[1][r]);
      mx = fmaxf(mx, __shfl_xor(mx, 1, 32));
      mx = fmaxf(mx, __shfl_xor(mx, 2, 32));
      mx = fmaxf(mx, __shfl_xor(mx, 4, 32));
      mx = fmaxf(mx, __shfl_xor(mx, 8, 32));
      const float mn = fmaxf(mi[r], mx);
      const float alpha = __expf(mi[r] - mn);
      const float p0 = __expf(s[0][r] - mn);
      const float p1 = __expf(s[1][r] - mn);
      float rs = p0 + p1;
      rs += __shfl_xor(rs, 1, 32);
      rs += __shfl_xor(rs, 2, 32);
      rs += __shfl_xor(rs, 4, 32);
      rs += __shfl_xor(rs, 8, 32);
      li[r] = li[r] * alpha + rs;
      mi[r] = mn;
      acc0[r] *= alpha; acc1[r] *= alpha; acc2[r] *= alpha; acc3[r] *= alpha;
      const int m = r + 8 * half;
      pbuf[wv][m][lo]      = (_Float16)p0;              // C-layout -> (m, k) in LDS
      pbuf[wv][m][16 + lo] = (_Float16)p1;
    }
    __builtin_amdgcn_wave_barrier();
    asm volatile("s_wait_dscnt 0" ::: "memory");        // LDS write->read, same wave
    const h16v pf = frag_load(&pbuf[wv][lo][0], 0, half);   // A-layout P (16x32)
    #pragma unroll
    for (int t = 0; t < 4; t++) {
      const h16v vf = frag_load(&vtile[buf][t * 16 + lo][0], 0, half);
      f8v& ac = (t == 0 ? acc0 : t == 1 ? acc1 : t == 2 ? acc2 : acc3);
      ac = wmma32(pf, vf, ac);
    }
    __syncthreads();                                    // done reading buf
  }

  #pragma unroll
  for (int r = 0; r < 8; r++) {
    const int m = r + 8 * half;
    const long token = (long)b * NTOK + qbase + m;
    const float inv = 1.0f / li[r];
    _Float16* o = aout + token * DIMC + hh * HD + lo;
    o[0]  = (_Float16)(acc0[r] * inv);
    o[16] = (_Float16)(acc1[r] * inv);
    o[32] = (_Float16)(acc2[r] * inv);
    o[48] = (_Float16)(acc3[r] * inv);
  }
}

// ---------------------------------------------------------------- proj GEMM
// out(4096 x 768) = A_f16 @ BT_f16^T + bias, fp32 output; 32x64 tile per wave.

__global__ void k_gemm_proj(const _Float16* __restrict__ A, const _Float16* __restrict__ BT,
                            const float* __restrict__ bias, float* __restrict__ out) {
  const int lane = threadIdx.x & 31, half = lane >> 4, lo = lane & 15;
  const int wv = threadIdx.x >> 5;
  const int tile = blockIdx.x * 8 + wv;
  const int TN = DIMC / 64;                 // 12 column strips
  const int tm = tile / TN, tn = tile % TN;

  const _Float16* arow0 = A + (long)(tm * 32 + lo) * DIMC;
  const _Float16* arow1 = A + (long)(tm * 32 + 16 + lo) * DIMC;
  const _Float16* brow[4];
  #pragma unroll
  for (int j = 0; j < 4; j++) brow[j] = BT + (long)(tn * 64 + j * 16 + lo) * DIMC;

  f8v acc[2][4] = {};
  #pragma unroll 2
  for (int kc = 0; kc < DIMC; kc += 32) {
    const h16v a0 = frag_load(arow0, kc, half);
    const h16v a1 = frag_load(arow1, kc, half);
    #pragma unroll
    for (int j = 0; j < 4; j++) {
      const h16v bf = frag_load(brow[j], kc, half);
      acc[0][j] = wmma32(a0, bf, acc[0][j]);
      acc[1][j] = wmma32(a1, bf, acc[1][j]);
    }
  }

  #pragma unroll
  for (int j = 0; j < 4; j++) {
    const int c = tn * 64 + j * 16 + lo;
    const float bs = bias[c];
    #pragma unroll
    for (int i = 0; i < 2; i++) {
      #pragma unroll
      for (int r = 0; r < 8; r++) {
        const int row = tm * 32 + i * 16 + r + 8 * half;
        out[(long)row * DIMC + c] = acc[i][j][r] + bs;
      }
    }
  }
}

// ---------------------------------------------------------------- launch

extern "C" void kernel_launch(void* const* d_in, const int* in_sizes, int n_in,
                              void* d_out, int out_size, void* d_ws, size_t ws_size,
                              hipStream_t stream) {
  const float* x      = (const float*)d_in[0];
  const float* qkv_w  = (const float*)d_in[1];
  const float* qkv_b  = (const float*)d_in[2];
  const float* proj_w = (const float*)d_in[3];
  const float* proj_b = (const float*)d_in[4];
  const float* rpd    = (const float*)d_in[5];
  const float* rph    = (const float*)d_in[6];
  const float* rpw    = (const float*)d_in[7];
  (void)in_sizes; (void)n_in; (void)out_size; (void)ws_size;

  char* ws = (char*)d_ws;
  size_t off = 0;
  auto carve = [&](size_t bytes) {
    void* p = ws + off;
    off += (bytes + 255) & ~(size_t)255;
    return p;
  };
  _Float16* xh   = (_Float16*)carve((size_t)ROWS * DIMC * 2);
  _Float16* wqT  = (_Float16*)carve((size_t)2304 * 768 * 2);
  _Float16* wpT  = (_Float16*)carve((size_t)768 * 768 * 2);
  _Float16* qws  = (_Float16*)carve((size_t)BH * NTOK * HD * 2);
  _Float16* kws  = (_Float16*)carve((size_t)BH * NTOK * HD * 2);
  _Float16* vTws = (_Float16*)carve((size_t)BH * NTOK * HD * 2);
  float*    rdw  = (float*)carve((size_t)BH * NTOK * 8 * 4);
  float*    rhw  = (float*)carve((size_t)BH * NTOK * 16 * 4);
  float*    rww  = (float*)carve((size_t)BH * NTOK * 16 * 4);
  _Float16* aout = (_Float16*)carve((size_t)ROWS * DIMC * 2);

  k_cvt<<<(ROWS * DIMC) / 256, 256, 0, stream>>>(x, xh, ROWS * DIMC);
  k_transpose<<<(768 * 2304 + 255) / 256, 256, 0, stream>>>(qkv_w, wqT, 768, 2304);
  k_transpose<<<(768 * 768 + 255) / 256, 256, 0, stream>>>(proj_w, wpT, 768, 768);
  k_gemm_qkv<<<576, 256, 0, stream>>>(xh, wqT, qkv_b, qws, kws, vTws);     // 4608/8 tiles
  k_rel<<<(BH * NTOK * 40 + 255) / 256, 256, 0, stream>>>(qws, rpd, rph, rpw, rdw, rhw, rww);
  k_attn<<<dim3(BH, NTOK / 128), 256, 0, stream>>>(qws, kws, vTws, rdw, rhw, rww, aout);
  k_gemm_proj<<<192, 256, 0, stream>>>(aout, wpT, proj_b, (float*)d_out);  // 1536/8 tiles
}
